// SwinCrackNet_37237366456517
// MI455X (gfx1250) — matrix operand order repaired and checked by reading, not compiled, not run on hardware
//
#include <hip/hip_runtime.h>
#include <hip/hip_bf16.h>
#include <math.h>

// ---------------------------------------------------------------------------
// Types for CDNA5 WMMA (wave32, 16x16x32 bf16 -> f32)
// ---------------------------------------------------------------------------
typedef __attribute__((ext_vector_type(16))) __bf16 v16bf;
typedef __attribute__((ext_vector_type(8)))  float  v8f;

union FragU { v16bf v; uint4 q[2]; };

__device__ __forceinline__ unsigned short f2bf(float f) {
  unsigned int u = __float_as_uint(f);
  u += 0x7fffu + ((u >> 16) & 1u);          // round-to-nearest-even
  return (unsigned short)(u >> 16);
}

__device__ __forceinline__ v8f wmma_bf16(v16bf a, v16bf b, v8f c) {
  return __builtin_amdgcn_wmma_f32_16x16x32_bf16(false, a, false, b,
                                                 (short)0, c, false, false);
}

// CDNA5 async global->LDS copy (16B per lane), tracked by ASYNCcnt.
// Low 32 bits of a flat LDS pointer are the LDS byte offset (ISA 10.2).
__device__ __forceinline__ void async_ld_b128(unsigned short* lds_dst,
                                              const unsigned short* gsrc) {
  unsigned loff = (unsigned)(size_t)lds_dst;
  asm volatile("global_load_async_to_lds_b128 %0, %1, off"
               :: "v"(loff), "v"(gsrc) : "memory");
}
__device__ __forceinline__ void wait_async0() {
  asm volatile("s_wait_asynccnt 0x0" ::: "memory");
}

// ---------------------------------------------------------------------------
// GEMM: C[M,N] = act(A[M,K](f32) @ B[N,K](bf16)^T + bias), bf16 WMMA core.
// 128x128 block tile, 4 waves each computing a 64x64 sub-tile of 16 WMMA frags.
// A tile: global_load_b128 (float4) + in-register f2bf. B tile: raw bf16 via
// global_load_async_to_lds_b128 + s_wait_asynccnt.
// ---------------------------------------------------------------------------
#define LDA_S 40   // ushorts per LDS row (80B, 16B aligned, bank-rotating)

__global__ __launch_bounds__(128) void gemm_bf16_wmma(
    const float* __restrict__ A, const unsigned short* __restrict__ Bb,
    const float* __restrict__ bias, float* __restrict__ C,
    int M, int N, int K, int act)
{
  __shared__ __align__(16) unsigned short As[128 * LDA_S];
  __shared__ __align__(16) unsigned short Bs[128 * LDA_S];

  const int tid  = threadIdx.x;
  const int lane = tid & 31;
  const int wave = tid >> 5;
  const int wm = wave >> 1, wn = wave & 1;
  const long m0 = (long)blockIdx.y * 128;
  const long n0 = (long)blockIdx.x * 128;

  const v8f vzero = {0.f,0.f,0.f,0.f,0.f,0.f,0.f,0.f};
  v8f acc[4][4];
  for (int i = 0; i < 4; ++i)
    for (int j = 0; j < 4; ++j) acc[i][j] = vzero;

  const long rA = m0 + tid;
  const long rB = n0 + tid;
  const int rr = lane & 15;
  const int hi = lane >> 4;          // 0 or 1
  const int koffA = hi * 8;          // A-frag: K chunks at hi*8 and hi*8+16
  const int koffB = hi * 8;          // B-frag: K chunks at hi*8 and hi*8+16

  for (int k0 = 0; k0 < K; k0 += 32) {
    __syncthreads();
    // ---- A tile: vector fp32 loads, convert to bf16 in registers ----
    {
      unsigned short* dA = &As[tid * LDA_S];
      if (rA < M) {
        const float* s = A + rA * K + k0;
#pragma unroll
        for (int c4 = 0; c4 < 8; ++c4) {
          const int kk = c4 * 4;
          if (k0 + kk + 4 <= K) {
            float4 v = *(const float4*)(s + kk);
            dA[kk]     = f2bf(v.x);
            dA[kk + 1] = f2bf(v.y);
            dA[kk + 2] = f2bf(v.z);
            dA[kk + 3] = f2bf(v.w);
          } else {
            for (int t = 0; t < 4; ++t)
              dA[kk + t] = (k0 + kk + t < K) ? f2bf(s[kk + t]) : (unsigned short)0;
          }
        }
      } else {
        for (int kk = 0; kk < 32; ++kk) dA[kk] = 0;
      }
    }
    // ---- B tile: raw bf16, async global->LDS (ASYNCcnt path) ----
    {
      unsigned short* dB = &Bs[tid * LDA_S];
      if (rB < N) {
        const unsigned short* s = Bb + rB * K + k0;
#pragma unroll
        for (int c8 = 0; c8 < 4; ++c8) {
          const int kk = c8 * 8;
          if (k0 + kk + 8 <= K) {
            async_ld_b128(dB + kk, s + kk);
          } else {
            for (int t = 0; t < 8; ++t)
              dB[kk + t] = (k0 + kk + t < K) ? s[kk + t] : (unsigned short)0;
          }
        }
      } else {
        for (int kk = 0; kk < 32; ++kk) dB[kk] = 0;
      }
    }
    wait_async0();
    __syncthreads();

    FragU af[4], bfj[4];
#pragma unroll
    for (int i = 0; i < 4; ++i) {
      const unsigned short* p = &As[(wm * 64 + i * 16 + rr) * LDA_S];
      af[i].q[0] = *(const uint4*)(p + koffA);
      af[i].q[1] = *(const uint4*)(p + koffA + 16);
    }
#pragma unroll
    for (int j = 0; j < 4; ++j) {
      const unsigned short* p = &Bs[(wn * 64 + j * 16 + rr) * LDA_S];
      bfj[j].q[0] = *(const uint4*)(p + koffB);
      bfj[j].q[1] = *(const uint4*)(p + koffB + 16);
    }
#pragma unroll
    for (int i = 0; i < 4; ++i)
#pragma unroll
      for (int j = 0; j < 4; ++j)
        acc[i][j] = wmma_bf16(af[i].v, bfj[j].v, acc[i][j]);
  }

#pragma unroll
  for (int i = 0; i < 4; ++i) {
    long rowBase = m0 + wm * 64 + i * 16 + hi * 8;
#pragma unroll
    for (int j = 0; j < 4; ++j) {
      long col = n0 + wn * 64 + j * 16 + rr;
      if (col < N) {
        float bv = bias ? bias[col] : 0.f;
#pragma unroll
        for (int r = 0; r < 8; ++r) {
          long row = rowBase + r;
          if (row < M) {
            float v = acc[i][j][r] + bv;
            if (act == 1)  // exact GELU
              v = 0.5f * v * (1.f + erff(v * 0.70710678118654752f));
            C[row * N + col] = v;
          }
        }
      }
    }
  }
}

// ---------------------------------------------------------------------------
// Fused windowed attention: one block per (batch, window, head).
// 49 tokens padded to 64. S = (Q*scale)@K^T (WMMA) + rel-pos bias + shift
// mask -> softmax -> O = P@V (WMMA). V stored transposed in LDS so both
// matmuls use identical contiguous b128 fragment loads.
// ---------------------------------------------------------------------------
#define AS_S 72  // ushorts per LDS row (144B, 16B aligned, bank-rotating)

__device__ __forceinline__ int band7(int g, int D, int shift) {
  return (g < D - 7) ? 0 : ((g < D - shift) ? 1 : 2);
}

__global__ __launch_bounds__(128) void win_attn_wmma(
    const float* __restrict__ qkv,   // [T, 3C]
    float* __restrict__ outp,        // [T, C]
    const float* __restrict__ rpb,   // [169, nh]
    int H, int W, int C, int nh, int hd, int shift)
{
  __shared__ __align__(16) unsigned short Qs[64 * AS_S];
  __shared__ __align__(16) unsigned short Ks[64 * AS_S];
  __shared__ __align__(16) unsigned short Vt[64 * AS_S];
  __shared__ __align__(16) unsigned short Ps[64 * AS_S];
  __shared__ float Ss[64 * 65];

  const int nWw = W / 7, nWh = H / 7;
  long bid = blockIdx.x;
  const int h  = (int)(bid % nh); bid /= nh;
  const int wx = (int)(bid % nWw); bid /= nWw;
  const int wy = (int)(bid % nWh); bid /= nWh;
  const long b = bid;
  const long tokBase = (((b * nWh + wy) * nWw) + wx) * 49;
  const int C3 = 3 * C;
  const float scale = rsqrtf((float)hd);

  const int tid = threadIdx.x;
  for (int i = tid; i < 64 * 64; i += 128) {
    int n = i >> 6, d = i & 63;
    unsigned short q = 0, k = 0, v = 0;
    if (n < 49 && d < hd) {
      const float* row = qkv + (tokBase + n) * C3;
      q = f2bf(row[h * hd + d] * scale);
      k = f2bf(row[C + h * hd + d]);
      v = f2bf(row[2 * C + h * hd + d]);
    }
    Qs[n * AS_S + d] = q;
    Ks[n * AS_S + d] = k;
    Vt[d * AS_S + n] = v;     // V transposed: row = head-dim, col = token
  }
  __syncthreads();

  const int lane = tid & 31;
  const int wave = tid >> 5;
  const int rr = lane & 15;
  const int hi = lane >> 4;
  const v8f vzero = {0.f,0.f,0.f,0.f,0.f,0.f,0.f,0.f};

  // S = Q @ K^T  (each wave: 16 rows x 64 cols)
  v8f sc[4];
  for (int j = 0; j < 4; ++j) sc[j] = vzero;
  const int kIter = (hd + 31) / 32;
  for (int k0 = 0; k0 < kIter; ++k0) {
    FragU af;
    const unsigned short* pa = &Qs[(wave * 16 + rr) * AS_S + k0 * 32];
    af.q[0] = *(const uint4*)(pa + hi * 8);
    af.q[1] = *(const uint4*)(pa + hi * 8 + 16);
#pragma unroll
    for (int j = 0; j < 4; ++j) {
      FragU bf_;
      const unsigned short* pb = &Ks[(j * 16 + rr) * AS_S + k0 * 32];
      bf_.q[0] = *(const uint4*)(pb + hi * 8);
      bf_.q[1] = *(const uint4*)(pb + hi * 8 + 16);
      sc[j] = wmma_bf16(af.v, bf_.v, sc[j]);
    }
  }
#pragma unroll
  for (int j = 0; j < 4; ++j)
#pragma unroll
    for (int r = 0; r < 8; ++r)
      Ss[(wave * 16 + hi * 8 + r) * 65 + j * 16 + rr] = sc[j][r];
  __syncthreads();

  // bias + mask + softmax (one thread per row)
  if (tid < 64) {
    const int i = tid;
    if (i < 49) {
      int iy = i / 7, ix = i % 7;
      int regI = 0;
      if (shift > 0)
        regI = band7(wy * 7 + iy, H, shift) * 3 + band7(wx * 7 + ix, W, shift);
      float vals[49];
      float mx = -1e30f;
      for (int j = 0; j < 49; ++j) {
        int jy = j / 7, jx = j % 7;
        float v = Ss[i * 65 + j];
        int rel = (iy - jy + 6) * 13 + (ix - jx + 6);
        v += rpb[rel * nh + h];
        if (shift > 0) {
          int regJ = band7(wy * 7 + jy, H, shift) * 3 + band7(wx * 7 + jx, W, shift);
          if (regJ != regI) v -= 100.f;
        }
        vals[j] = v;
        mx = fmaxf(mx, v);
      }
      float s = 0.f;
      for (int j = 0; j < 49; ++j) { vals[j] = expf(vals[j] - mx); s += vals[j]; }
      float inv = 1.f / s;
      for (int j = 0; j < 49; ++j) Ps[i * AS_S + j] = f2bf(vals[j] * inv);
      for (int j = 49; j < 64; ++j) Ps[i * AS_S + j] = 0;
    } else {
      for (int j = 0; j < 64; ++j) Ps[i * AS_S + j] = 0;
    }
  }
  __syncthreads();

  // O = P @ V
  v8f oc[4];
  for (int j = 0; j < 4; ++j) oc[j] = vzero;
  for (int k0 = 0; k0 < 2; ++k0) {
    FragU af;
    const unsigned short* pa = &Ps[(wave * 16 + rr) * AS_S + k0 * 32];
    af.q[0] = *(const uint4*)(pa + hi * 8);
    af.q[1] = *(const uint4*)(pa + hi * 8 + 16);
#pragma unroll
    for (int j = 0; j < 4; ++j) {
      FragU bf_;
      const unsigned short* pb = &Vt[(j * 16 + rr) * AS_S + k0 * 32];
      bf_.q[0] = *(const uint4*)(pb + hi * 8);
      bf_.q[1] = *(const uint4*)(pb + hi * 8 + 16);
      oc[j] = wmma_bf16(af.v, bf_.v, oc[j]);
    }
  }
#pragma unroll
  for (int j = 0; j < 4; ++j) {
    int d = j * 16 + rr;
    if (d < hd) {
#pragma unroll
      for (int r = 0; r < 8; ++r) {
        int row = wave * 16 + hi * 8 + r;
        if (row < 49)
          outp[(tokBase + row) * C + h * hd + d] = oc[j][r];
      }
    }
  }
}

// ---------------------------------------------------------------------------
// LayerNorm: one block (128 threads) per row.
// ---------------------------------------------------------------------------
__global__ __launch_bounds__(128) void layernorm_k(
    const float* __restrict__ x, const float* __restrict__ g,
    const float* __restrict__ b, float* __restrict__ y, int C)
{
  __shared__ float red[128], red2[128];
  const long row = blockIdx.x;
  const float* xr = x + row * C;
  float s = 0.f, s2 = 0.f;
  for (int c = threadIdx.x; c < C; c += 128) { float v = xr[c]; s += v; s2 += v * v; }
  red[threadIdx.x] = s; red2[threadIdx.x] = s2;
  __syncthreads();
  for (int off = 64; off > 0; off >>= 1) {
    if (threadIdx.x < off) {
      red[threadIdx.x]  += red[threadIdx.x + off];
      red2[threadIdx.x] += red2[threadIdx.x + off];
    }
    __syncthreads();
  }
  float mean = red[0] / C;
  float var  = red2[0] / C - mean * mean;
  float inv  = rsqrtf(var + 1e-5f);
  float* yr = y + row * C;
  for (int c = threadIdx.x; c < C; c += 128)
    yr[c] = (xr[c] - mean) * inv * g[c] + b[c];
}

// ---------------------------------------------------------------------------
// Elementwise / gather kernels (grid-stride)
// ---------------------------------------------------------------------------
#define GS_LOOP(i, n) \
  for (long i = (long)blockIdx.x * 256 + threadIdx.x, _st = (long)gridDim.x * 256; \
       i < (n); i += _st)

__global__ void k_cvt_bf16(const float* __restrict__ x,
                           unsigned short* __restrict__ y, long n) {
  GS_LOOP(i, n) y[i] = f2bf(x[i]);
}

__global__ void k_add(const float* __restrict__ a, const float* __restrict__ b,
                      float* __restrict__ y, long n) {
  GS_LOOP(i, n) y[i] = a[i] + b[i];
}

// [B,H,W,C] (with optional cyclic shift) -> windowed [B*nW*49, C]
__global__ void k_win_gather(const float* __restrict__ x, float* __restrict__ out,
                             int B, int H, int W, int C, int shift, long n) {
  const int nWw = W / 7, nWh = H / 7; (void)B;
  GS_LOOP(i, n) {
    long t = i / C; int c = (int)(i - t * C);
    int loc = (int)(t % 49); long wn = t / 49;
    int wx = (int)(wn % nWw); wn /= nWw;
    int wy = (int)(wn % nWh); long b = wn / nWh;
    int y = loc / 7, xx = loc % 7;
    int gy = wy * 7 + y, gx = wx * 7 + xx;
    if (shift) { gy = (gy + shift) % H; gx = (gx + shift) % W; }
    out[i] = x[((b * H + gy) * (long)W + gx) * C + c];
  }
}

// windowed [B*nW*49, C] -> [B,H,W,C] with reverse shift, + residual
__global__ void k_win_scatter_add(const float* __restrict__ xin,
                                  const float* __restrict__ win,
                                  float* __restrict__ xout,
                                  int B, int H, int W, int C, int shift, long n) {
  const int nWw = W / 7, nWh = H / 7; (void)B;
  GS_LOOP(i, n) {
    long t = i / C; int c = (int)(i - t * C);
    int gx = (int)(t % W); long p = t / W;
    int gy = (int)(p % H); long b = p / H;
    int ry = gy, rx = gx;
    if (shift) { ry = (gy - shift + H) % H; rx = (gx - shift + W) % W; }
    int wy = ry / 7, y = ry % 7, wx = rx / 7, xx = rx % 7;
    long wt = ((b * nWh + wy) * (long)nWw + wx) * 49 + y * 7 + xx;
    xout[i] = xin[i] + win[wt * C + c];
  }
}

// 2x2 patch-merge concat: [B,H,W,C] -> [B,(H/2)*(W/2),4C]
__global__ void k_merge_gather(const float* __restrict__ x, float* __restrict__ out,
                               int B, int H, int W, int C, long n) {
  const int H2 = H / 2, W2 = W / 2, C4 = 4 * C; (void)B;
  GS_LOOP(i, n) {
    long t = i / C4; int cc = (int)(i - t * C4);
    int w2 = (int)(t % W2); long p = t / W2;
    int h2 = (int)(p % H2); long b = p / H2;
    int q = cc / C, c = cc - q * C;
    int dy = q & 1, dx = q >> 1;   // order: (0,0),(1,0),(0,1),(1,1)
    out[i] = x[((b * H + 2 * h2 + dy) * (long)W + 2 * w2 + dx) * C + c];
  }
}

// pixel-shuffle expand: in [B,H*W,Cin] -> out [B,(f*H)*(f*W),Cin/f^2]
__global__ void k_expand_rearrange(const float* __restrict__ in, float* __restrict__ out,
                                   int B, int H, int W, int Cin, int f, long n) {
  const int Cout = Cin / (f * f);
  const int fW = f * W, fH = f * H; (void)B;
  GS_LOOP(i, n) {
    long t = i / Cout; int cc = (int)(i - t * Cout);
    long p = t / fW; int X = (int)(t - p * fW);
    long b = p / fH; int Y = (int)(p - b * fH);
    int hy = Y / f, r = Y % f, wx = X / f, s = X % f;
    out[i] = in[((b * H + hy) * (long)W + wx) * Cin + (long)(r * f + s) * Cout + cc];
  }
}

__global__ void k_concat(const float* __restrict__ a, const float* __restrict__ b,
                         float* __restrict__ out, int Ca, int Cb, long n) {
  const int Ct = Ca + Cb;
  GS_LOOP(i, n) {
    long r = i / Ct; int c = (int)(i - r * Ct);
    out[i] = (c < Ca) ? a[r * (long)Ca + c] : b[r * (long)Cb + (c - Ca)];
  }
}

// im2col for 4x4/stride-4 patch embed: [B,3,224,224] -> [B*56*56, 48]
__global__ void k_im2col(const float* __restrict__ x, float* __restrict__ col, long n) {
  GS_LOOP(i, n) {
    long r = i / 48; int k = (int)(i - r * 48);
    int px = (int)(r % 56); long t = r / 56;
    int py = (int)(t % 56); long b = t / 56;
    int ci = k >> 4, kk = k & 15, ky = kk >> 2, kx = kk & 3;
    col[i] = x[((b * 3 + ci) * 224 + (py * 4 + ky)) * (long)224 + (px * 4 + kx)];
  }
}

// out[b,o,h,w] = sum_c x[b,(h,w),c] * w[o,c]
__global__ void k_outproj(const float* __restrict__ x, const float* __restrict__ w,
                          float* __restrict__ out, long n, int HW, int C) {
  GS_LOOP(i, n) {
    int p = (int)(i % HW); long t = i / HW;
    int o = (int)(t % 2); long b = t / 2;
    const float* xr = x + (b * HW + p) * (long)C;
    const float* wr = w + o * C;
    float s = 0.f;
    for (int c = 0; c < C; ++c) s += xr[c] * wr[c];
    out[i] = s;
  }
}

// ---------------------------------------------------------------------------
// Host orchestration
// ---------------------------------------------------------------------------
extern "C" void kernel_launch(void* const* d_in, const int* in_sizes, int n_in,
                              void* d_out, int out_size, void* d_ws, size_t ws_size,
                              hipStream_t stream) {
  (void)in_sizes; (void)n_in; (void)out_size; (void)ws_size;
  const int B = 4;
  int cur = 0;
  auto nxt = [&]() -> const float* { return (const float*)d_in[cur++]; };

  // Parameter pointers in jax pytree order (dict keys sorted alphabetically):
  // params{down, expand0, ffns, final, norm_down, norm_up, out_w, patch_embed, up}, x
  struct BlockP {
    const float *fc1_b,*fc1_w,*fc2_b,*fc2_w,*n1b,*n1g,*n2b,*n2g,
                *proj_b,*proj_w,*qkv_b,*qkv_w,*rpb;
  };
  auto readBlock = [&]() {
    BlockP p;
    p.fc1_b=nxt(); p.fc1_w=nxt(); p.fc2_b=nxt(); p.fc2_w=nxt();
    p.n1b=nxt();   p.n1g=nxt();   p.n2b=nxt();   p.n2g=nxt();
    p.proj_b=nxt();p.proj_w=nxt();p.qkv_b=nxt(); p.qkv_w=nxt(); p.rpb=nxt();
    return p;
  };

  const int DEPTHS_[4]  = {2, 2, 6, 2};
  const int HEADS_[4]   = {3, 6, 12, 24};
  const int DE_HEADS_[3]= {16, 8, 4};

  BlockP dblk[4][6];
  const float *m_lnb[3], *m_lng[3], *m_w[3];
  for (int i = 0; i < 4; ++i) {
    for (int d = 0; d < DEPTHS_[i]; ++d) dblk[i][d] = readBlock();
    if (i < 3) { m_lnb[i]=nxt(); m_lng[i]=nxt(); m_w[i]=nxt(); }
  }
  const float *e0_lnb=nxt(), *e0_lng=nxt(), *e0_w=nxt();
  const float *f_fc1b[3], *f_fc1w[3], *f_fc2b[3], *f_fc2w[3], *f_lnb[3], *f_lng[3];
  for (int i = 0; i < 3; ++i) {
    f_fc1b[i]=nxt(); f_fc1w[i]=nxt(); f_fc2b[i]=nxt(); f_fc2w[i]=nxt();
    f_lnb[i]=nxt();  f_lng[i]=nxt();
  }
  const float *fin_lnb=nxt(), *fin_lng=nxt(), *fin_w=nxt();
  const float *nd_b=nxt(), *nd_g=nxt();
  const float *nu_b=nxt(), *nu_g=nxt();
  const float *out_w=nxt();
  const float *pe_b=nxt(), *pe_lnb=nxt(), *pe_lng=nxt(), *pe_w=nxt();
  BlockP ublk[3][2];
  const float *u_elnb[2], *u_elng[2], *u_ew[2], *u_rb[2], *u_rw[2];
  for (int j = 0; j < 3; ++j) {
    for (int d = 0; d < 2; ++d) ublk[j][d] = readBlock();
    if (j < 2) { u_elnb[j]=nxt(); u_elng[j]=nxt(); u_ew[j]=nxt(); u_rb[j]=nxt(); u_rw[j]=nxt(); }
  }
  const float* x_in = nxt();

  // -------- workspace (bump allocator, float units) --------
  size_t off = 0;
  auto alloc = [&](size_t nfloat) -> float* {
    float* p = (float*)((char*)d_ws + off);
    off += ((nfloat * sizeof(float)) + 255) & ~(size_t)255;
    return p;
  };
  float* bufP = alloc(40u * 1000 * 1000);
  float* bufQ = alloc(40u * 1000 * 1000);
  float* t0 = alloc(10u * 1000 * 1000);
  float* t1 = alloc(10u * 1000 * 1000);
  float* t2 = alloc(10u * 1000 * 1000);
  float* t3 = alloc(10u * 1000 * 1000);
  float* xc = alloc(4u * 1000 * 1000);
  float* xa = alloc(4u * 1000 * 1000);
  unsigned short* wbuf = (unsigned short*)alloc(1300u * 1000);  // bf16 weights
  float* skip[3];
  skip[0] = alloc(1300u * 1000);
  skip[1] = alloc(700u * 1000);
  skip[2] = alloc(350u * 1000);

  auto gs = [](long n) {
    long blk = (n + 255) / 256;
    if (blk > 1048576) blk = 1048576;
    return dim3((unsigned)blk);
  };
  // Convert weight to bf16 once per call, then WMMA GEMM with async B staging.
  auto gemm = [&](const float* A, const float* Bm, const float* bias, float* Cp,
                  long M, int N, int K, int act) {
    long nw = (long)N * K;
    k_cvt_bf16<<<gs(nw), 256, 0, stream>>>(Bm, wbuf, nw);
    dim3 g((unsigned)((N + 127) / 128), (unsigned)((M + 127) / 128));
    gemm_bf16_wmma<<<g, dim3(128), 0, stream>>>(A, wbuf, bias, Cp, (int)M, N, K, act);
  };
  auto lnorm = [&](const float* x, const float* g, const float* b, float* y,
                   long R, int C) {
    layernorm_k<<<dim3((unsigned)R), dim3(128), 0, stream>>>(x, g, b, y, C);
  };
  auto dcopy = [&](float* dst, const float* src, long nfloat) {
    hipMemcpyAsync(dst, src, nfloat * sizeof(float), hipMemcpyDeviceToDevice, stream);
  };
  auto swap_x = [&]() { float* t = xc; xc = xa; xa = t; };

  auto swin = [&](int H, int W, int C, int nh, int shift, const BlockP& p) {
    long R = (long)B * H * W;
    long n = R * C;
    lnorm(xc, p.n1g, p.n1b, t0, R, C);
    k_win_gather<<<gs(n), 256, 0, stream>>>(t0, t1, B, H, W, C, shift, n);
    gemm(t1, p.qkv_w, p.qkv_b, t2, R, 3 * C, C, 0);
    int nWh = H / 7, nWw = W / 7;
    win_attn_wmma<<<dim3((unsigned)(B * nWh * nWw * nh)), dim3(128), 0, stream>>>(
        t2, t3, p.rpb, H, W, C, nh, C / nh, shift);
    gemm(t3, p.proj_w, p.proj_b, t1, R, C, C, 0);
    k_win_scatter_add<<<gs(n), 256, 0, stream>>>(xc, t1, xa, B, H, W, C, shift, n);
    swap_x();
    lnorm(xc, p.n2g, p.n2b, t0, R, C);
    gemm(t0, p.fc1_w, p.fc1_b, t2, R, 4 * C, C, 1);
    gemm(t2, p.fc2_w, p.fc2_b, t1, R, C, 4 * C, 0);
    k_add<<<gs(n), 256, 0, stream>>>(xc, t1, xc, n);
  };

  // -------- patch embed --------
  int H = 56, W = 56, C = 96;
  {
    long n = (long)B * 3136 * 48;
    k_im2col<<<gs(n), 256, 0, stream>>>(x_in, t0, n);
    gemm(t0, pe_w, pe_b, xc, (long)B * 3136, 96, 48, 0);
    lnorm(xc, pe_lng, pe_lnb, xc, (long)B * 3136, 96);
  }

  // -------- encoder --------
  for (int i = 0; i < 4; ++i) {
    long R = (long)B * H * W;
    if (i < 3) dcopy(skip[i], xc, R * C);
    for (int d = 0; d < DEPTHS_[i]; ++d) {
      int shift = (d % 2 == 1 && H > 7 && W > 7) ? 3 : 0;
      swin(H, W, C, HEADS_[i], shift, dblk[i][d]);
    }
    if (i < 3) {
      long Ro = R / 4;
      long n = Ro * 4 * C;
      k_merge_gather<<<gs(n), 256, 0, stream>>>(xc, t0, B, H, W, C, n);
      lnorm(t0, m_lng[i], m_lnb[i], t0, Ro, 4 * C);
      gemm(t0, m_w[i], nullptr, xa, Ro, 2 * C, 4 * C, 0);
      swap_x();
      H /= 2; W /= 2; C *= 2;
    }
  }
  lnorm(xc, nd_g, nd_b, xc, (long)B * H * W, C);

  // -------- skip FFNs --------
  {
    const int Cs[3] = {96, 192, 384};
    const int Ls[3] = {3136, 784, 196};
    for (int i = 0; i < 3; ++i) {
      long R = (long)B * Ls[i]; int Ci = Cs[i];
      lnorm(skip[i], f_lng[i], f_lnb[i], t0, R, Ci);
      gemm(t0, f_fc1w[i], f_fc1b[i], t2, R, 4 * Ci, Ci, 1);
      gemm(t2, f_fc2w[i], f_fc2b[i], t1, R, Ci, 4 * Ci, 0);
      k_add<<<gs(R * Ci), 256, 0, stream>>>(skip[i], t1, skip[i], R * Ci);
    }
  }

  // -------- expand0: 7x7x768 -> 14x14x384 --------
  {
    long R = (long)B * H * W;
    gemm(xc, e0_w, nullptr, t0, R, 2 * C, C, 0);
    long n = R * 2 * C;
    k_expand_rearrange<<<gs(n), 256, 0, stream>>>(t0, xa, B, H, W, 2 * C, 2, n);
    swap_x();
    H *= 2; W *= 2; C /= 2;
    lnorm(xc, e0_lng, e0_lnb, xc, (long)B * H * W, C);
  }

  // -------- decoder --------
  for (int j = 0; j < 3; ++j) {
    long R = (long)B * H * W;
    const float* sk = skip[2 - j];           // skip channels == current C
    long n = R * (2 * C);
    k_concat<<<gs(n), 256, 0, stream>>>(xc, sk, t0, C, C, n);
    if (j < 2) {
      gemm(t0, u_rw[j], u_rb[j], xa, R, C, 2 * C, 0);
      swap_x();
    } else {
      C = 2 * C;                              // 192, no reduce in last stage
      dcopy(xc, t0, R * C);
    }
    for (int d = 0; d < 2; ++d) {
      int shift = (d % 2 == 1 && H > 7 && W > 7) ? 3 : 0;
      swin(H, W, C, DE_HEADS_[j], shift, ublk[j][d]);
    }
    if (j < 2) {
      gemm(xc, u_ew[j], nullptr, t0, R, 2 * C, C, 0);
      long ne = R * 2 * C;
      k_expand_rearrange<<<gs(ne), 256, 0, stream>>>(t0, xa, B, H, W, 2 * C, 2, ne);
      swap_x();
      H *= 2; W *= 2; C /= 2;
      lnorm(xc, u_elng[j], u_elnb[j], xc, (long)B * H * W, C);
    }
  }
  lnorm(xc, nu_g, nu_b, xc, (long)B * H * W, C);   // H=W=56, C=192

  // -------- final 4x expansion + output projection --------
  {
    long R = (long)B * H * W;                 // 12544
    gemm(xc, fin_w, nullptr, bufP, R, 16 * C, C, 0);
    long n = R * 16 * C;
    k_expand_rearrange<<<gs(n), 256, 0, stream>>>(bufP, bufQ, B, H, W, 16 * C, 4, n);
    long R2 = R * 16;
    lnorm(bufQ, fin_lng, fin_lnb, bufQ, R2, C);
    long HW = (long)(4 * H) * (4 * W);
    long no = (long)B * 2 * HW;
    k_outproj<<<gs(no), 256, 0, stream>>>(bufQ, out_w, (float*)d_out, no, (int)HW, C);
  }
}
